// HyperbolicMemory_72919954751570
// MI455X (gfx1250) — matrix operand order repaired and checked by reading, not compiled, not run on hardware
//
#include <hip/hip_runtime.h>
#include <hip/hip_bf16.h>

#define BQ   1024
#define NMEM 50000
#define DD   256
#define MAXA 200
#define KK   8
#define NCHUNKS 98
#define CHUNK_COLS 512

typedef __attribute__((ext_vector_type(16))) __bf16 v16bf;
typedef __attribute__((ext_vector_type(8)))  __bf16 v8bf;
typedef __attribute__((ext_vector_type(8)))  float  v8f;

static __device__ __forceinline__ __bf16 f2bf(float f) {
  unsigned u = __builtin_bit_cast(unsigned, f);
  u += 0x7FFFu + ((u >> 16) & 1u);                 // round-to-nearest-even
  unsigned short h = (unsigned short)(u >> 16);
  return __builtin_bit_cast(__bf16, h);
}

// ---------------- W -> Wt (bf16, transposed so B-operand loads are contiguous)
__global__ void wt_kernel(const float* __restrict__ W, __bf16* __restrict__ Wt) {
  int i = blockIdx.x * blockDim.x + threadIdx.x;
  if (i < DD * DD) {
    int k = i >> 8, n = i & 255;
    Wt[n * DD + k] = f2bf(W[i]);
  }
}

// ---------------- H = rescale(tanh(X @ W + b)); one wave handles 16 rows x 256 cols
__global__ void project_kernel(const float* __restrict__ X, const __bf16* __restrict__ Wt,
                               const float* __restrict__ bias,
                               __bf16* __restrict__ H, float* __restrict__ rowsq) {
  __shared__ float Ht[16][DD + 4];
  __shared__ float rscale[16];
  const int lane = threadIdx.x;             // 0..31 (wave32)
  const int rowbase = blockIdx.x * 16;
  const int sub = lane & 15;
  const int hi  = lane >> 4;

  for (int ct = 0; ct < 16; ++ct) {
    const int ncol = ct * 16;
    v8f acc = {};
    #pragma unroll
    for (int kb = 0; kb < DD; kb += 32) {
      // A 16x32 bf16: lane<16 holds row=sub, K = kb + 0..15; lane>=16 shifts K by 16
      const float* arow = X + (size_t)(rowbase + sub) * DD + kb + hi * 16;
      v16bf a;
      #pragma unroll
      for (int e = 0; e < 8; ++e) a[e]     = f2bf(arow[e]);
      #pragma unroll
      for (int e = 0; e < 8; ++e) a[8 + e] = f2bf(arow[8 + e]);
      // B 32x16 bf16: lane holds column n = ncol+sub, K = kb + hi*16 + 0..15 (contiguous in Wt)
      v16bf bm = *(const v16bf*)(Wt + (size_t)(ncol + sub) * DD + kb + hi * 16);
      acc = __builtin_amdgcn_wmma_f32_16x16x32_bf16(false, a, false, bm,
                                                    (short)0, acc, false, false);
    }
    float bv = bias[ncol + sub];
    #pragma unroll
    for (int r = 0; r < 8; ++r)
      Ht[r + hi * 8][ncol + sub] = tanhf(acc[r] + bv);
  }
  __syncthreads();
  {
    int row = lane >> 1, half = lane & 1;
    float s = 0.f;
    for (int c = half * 128; c < half * 128 + 128; ++c) { float v = Ht[row][c]; s += v * v; }
    s += __shfl_xor(s, 1, 32);
    float norm  = sqrtf(s);
    float scale = (norm > 0.95f) ? (0.95f / norm) : 1.0f;
    if (half == 0) {
      rscale[row] = scale;
      rowsq[rowbase + row] = s * scale * scale;   // ||h||^2 after rescale
    }
  }
  __syncthreads();
  for (int i = 0; i < 128; ++i) {
    int e = lane + 32 * i;
    int row = e >> 8, col = e & 255;
    H[(size_t)(rowbase + row) * DD + col] = f2bf(Ht[row][col] * rscale[row]);
  }
}

// ---------------- pairwise arg + running top-8 per query row; one wave per (16q x 512m) chunk
__global__ void dist_topk_kernel(const __bf16* __restrict__ Hq, const __bf16* __restrict__ Hm,
                                 const float* __restrict__ qsq, const float* __restrict__ msq,
                                 float* __restrict__ argp, int* __restrict__ idxp) {
  __shared__ float argT[16][17];
  __shared__ float candV[16][16];
  __shared__ int   candI[16][16];
  const int lane  = threadIdx.x;
  const int qbase = blockIdx.x * 16;
  const int chunk = blockIdx.y;
  const int cstart = chunk * CHUNK_COLS;
  const int sub = lane & 15;
  const int hi  = lane >> 4;

  float qs[8];
  #pragma unroll
  for (int r = 0; r < 8; ++r) qs[r] = qsq[qbase + hi * 8 + r];

  float best[KK]; int bid[KK];
  #pragma unroll
  for (int j = 0; j < KK; ++j) { best[j] = 3.0e38f; bid[j] = 0; }

  int ntiles = (NMEM - cstart + 15) / 16;
  if (ntiles > CHUNK_COLS / 16) ntiles = CHUNK_COLS / 16;

  for (int t = 0; t < ntiles; ++t) {
    const int mbase  = cstart + t * 16;
    const int mcol   = mbase + sub;
    const int mclamp = mcol < NMEM ? mcol : NMEM - 1;
    // prefetch next tile's B rows into cache (global_prefetch_b8)
    if (t + 1 < ntiles) {
      int pn = mbase + 16 + sub; if (pn >= NMEM) pn = NMEM - 1;
      __builtin_prefetch(Hm + (size_t)pn * DD, 0, 0);
    }
    v8f acc = {};
    #pragma unroll
    for (int kb = 0; kb < DD; kb += 32) {
      const __bf16* ap = Hq + (size_t)(qbase + sub) * DD + kb + hi * 16;
      v8bf c0 = *(const v8bf*)(ap);
      v8bf c1 = *(const v8bf*)(ap + 8);
      v16bf a = __builtin_shufflevector(c0, c1, 0,1,2,3,4,5,6,7,8,9,10,11,12,13,14,15);
      v16bf bm = *(const v16bf*)(Hm + (size_t)mclamp * DD + kb + hi * 16);
      acc = __builtin_amdgcn_wmma_f32_16x16x32_bf16(false, a, false, bm,
                                                    (short)0, acc, false, false);
    }
    const float ms = msq[mclamp];
    const bool oob = (mcol >= NMEM);
    #pragma unroll
    for (int r = 0; r < 8; ++r) {
      float q = qs[r];
      float diff = fmaxf(q + ms - 2.0f * acc[r], 0.0f);
      float denom = (1.0f - q) * (1.0f - ms);
      float arg = 1.0f + 2.0f * diff / (denom + 1e-8f);
      arg = fmaxf(arg, 1.0f + 1e-6f);              // arccosh(arg) monotone -> rank by arg
      argT[r + hi * 8][sub] = oob ? 3.0e38f : arg;
    }
    __syncthreads();
    {
      int srow = lane >> 1;
      int cb   = (lane & 1) * 8;                   // 2 lanes per row, 8 cols each
      #pragma unroll
      for (int c = 0; c < 8; ++c) {
        float d = argT[srow][cb + c];
        if (d < best[KK - 1]) {
          float cv = d; int ci = mbase + cb + c;
          #pragma unroll
          for (int j = 0; j < KK; ++j) {
            if (cv < best[j]) {
              float tv = best[j]; int ti = bid[j];
              best[j] = cv; bid[j] = ci; cv = tv; ci = ti;
            }
          }
        }
      }
    }
    __syncthreads();
  }
  // merge the two lanes of each row -> 8 per row per chunk
  {
    int srow = lane >> 1, half = lane & 1;
    #pragma unroll
    for (int j = 0; j < KK; ++j) { candV[srow][half * 8 + j] = best[j]; candI[srow][half * 8 + j] = bid[j]; }
  }
  __syncthreads();
  if (lane < 16) {
    float fb[KK]; int fi[KK];
    #pragma unroll
    for (int j = 0; j < KK; ++j) { fb[j] = 3.0e38f; fi[j] = 0; }
    for (int c = 0; c < 16; ++c) {
      float cv = candV[lane][c]; int ci = candI[lane][c];
      if (cv < fb[KK - 1]) {
        #pragma unroll
        for (int j = 0; j < KK; ++j) {
          if (cv < fb[j]) { float tv = fb[j]; int ti = fi[j]; fb[j] = cv; fi[j] = ci; cv = tv; ci = ti; }
        }
      }
    }
    size_t base = ((size_t)(qbase + lane) * NCHUNKS + chunk) * KK;
    #pragma unroll
    for (int j = 0; j < KK; ++j) { argp[base + j] = fb[j]; idxp[base + j] = fi[j]; }
  }
}

// ---------------- merge partials, softmax weights, gather mask rows
__global__ void merge_kernel(const float* __restrict__ argp, const int* __restrict__ idxp,
                             const float* __restrict__ masks, float* __restrict__ out) {
  __shared__ float sv[256][KK];
  __shared__ int   si[256][KK];
  __shared__ int   fidx[KK];
  const int row = blockIdx.x;
  const int tid = threadIdx.x;
  const int NC  = NCHUNKS * KK;                    // 784 candidates
  const float* av = argp + (size_t)row * NC;
  const int*   ai = idxp + (size_t)row * NC;

  float best[KK]; int bid[KK];
  #pragma unroll
  for (int j = 0; j < KK; ++j) { best[j] = 3.0e38f; bid[j] = 0; }
  for (int c = tid; c < NC; c += 256) {
    float cv = av[c]; int ci = ai[c];
    if (cv < best[KK - 1]) {
      #pragma unroll
      for (int j = 0; j < KK; ++j)
        if (cv < best[j]) { float tv = best[j]; int ti = bid[j]; best[j] = cv; bid[j] = ci; cv = tv; ci = ti; }
    }
  }
  #pragma unroll
  for (int j = 0; j < KK; ++j) { sv[tid][j] = best[j]; si[tid][j] = bid[j]; }
  __syncthreads();
  for (int s = 128; s >= 1; s >>= 1) {
    if (tid < s) {
      #pragma unroll
      for (int j = 0; j < KK; ++j) {
        float cv = sv[tid + s][j]; int ci = si[tid + s][j];
        if (cv < best[KK - 1]) {
          #pragma unroll
          for (int q = 0; q < KK; ++q)
            if (cv < best[q]) { float tv = best[q]; int ti = bid[q]; best[q] = cv; bid[q] = ci; cv = tv; ci = ti; }
        }
      }
      #pragma unroll
      for (int j = 0; j < KK; ++j) { sv[tid][j] = best[j]; si[tid][j] = bid[j]; }
    }
    __syncthreads();
  }
  if (tid == 0) {
    float dist[KK]; float dmin = 3.0e38f;
    #pragma unroll
    for (int j = 0; j < KK; ++j) {
      float a = best[j];
      dist[j] = logf(a + sqrtf(fmaxf(a * a - 1.0f, 0.0f)));   // arccosh
      dmin = fminf(dmin, dist[j]);
    }
    float w[KK]; float sum = 0.f;
    #pragma unroll
    for (int j = 0; j < KK; ++j) { w[j] = expf(-(dist[j] - dmin)); sum += w[j]; }
    float inv = 1.0f / sum;
    #pragma unroll
    for (int j = 0; j < KK; ++j) {
      out[(size_t)row * KK + j] = w[j] * inv;
      fidx[j] = bid[j];
    }
  }
  __syncthreads();
  float* hints = out + (size_t)BQ * KK;
  #pragma unroll
  for (int j = 0; j < KK; ++j) {
    int mi = fidx[j];
    for (int t = tid; t < MAXA; t += 256)
      hints[((size_t)row * KK + j) * MAXA + t] = masks[(size_t)mi * MAXA + t];
  }
}

extern "C" void kernel_launch(void* const* d_in, const int* in_sizes, int n_in,
                              void* d_out, int out_size, void* d_ws, size_t ws_size,
                              hipStream_t stream) {
  const float* Q     = (const float*)d_in[0];   // [B,D]
  const float* M     = (const float*)d_in[1];   // [N,D]
  const float* masks = (const float*)d_in[2];   // [N,MAX_ATOMS]
  const float* W     = (const float*)d_in[3];   // [D,D]
  const float* bias  = (const float*)d_in[4];   // [D]
  (void)in_sizes; (void)n_in; (void)out_size; (void)ws_size; // k is compile-time KK

  char* ws = (char*)d_ws;
  size_t off = 0;
  auto alloc = [&](size_t bytes) -> void* {
    void* p = ws + off;
    off = (off + bytes + 255) & ~(size_t)255;
    return p;
  };
  __bf16* Wt  = (__bf16*)alloc((size_t)DD * DD * sizeof(__bf16));
  __bf16* Hq  = (__bf16*)alloc((size_t)BQ * DD * sizeof(__bf16));
  __bf16* Hm  = (__bf16*)alloc((size_t)NMEM * DD * sizeof(__bf16));
  float*  qsq = (float*)alloc((size_t)BQ * sizeof(float));
  float*  msq = (float*)alloc((size_t)NMEM * sizeof(float));
  float*  argp = (float*)alloc((size_t)BQ * NCHUNKS * KK * sizeof(float));
  int*    idxp = (int*)alloc((size_t)BQ * NCHUNKS * KK * sizeof(int));

  wt_kernel<<<(DD * DD + 255) / 256, 256, 0, stream>>>(W, Wt);
  project_kernel<<<BQ / 16,   32, 0, stream>>>(Q, Wt, bias, Hq, qsq);
  project_kernel<<<NMEM / 16, 32, 0, stream>>>(M, Wt, bias, Hm, msq);
  dist_topk_kernel<<<dim3(BQ / 16, NCHUNKS), 32, 0, stream>>>(Hq, Hm, qsq, msq, argp, idxp);
  merge_kernel<<<BQ, 256, 0, stream>>>(argp, idxp, masks, (float*)d_out);
}
